// EGConvNetwork_76828374991622
// MI455X (gfx1250) — matrix operand order repaired and checked by reading, not compiled, not run on hardware
//
#include <hip/hip_runtime.h>

#define NIN   128
#define NHID  256
#define NB    4
#define NH    8
#define NFH   32
#define NG    512
#define NL    4

typedef __attribute__((ext_vector_type(16))) _Float16 v16h;
typedef __attribute__((ext_vector_type(8)))  float    v8f;

__device__ __forceinline__ void atomAddF(float* p, float v) {
  __hip_atomic_fetch_add(p, v, __ATOMIC_RELAXED, __HIP_MEMORY_SCOPE_AGENT);
}

// ---------------------------------------------------------------------------
// Weight pre-pack: f32 W[kin,kout] -> f16x2 dwords laid out exactly as the
// WMMA B-fragment register image.  dword index = ((c*nkt + kt)*32 + lane)*8 + v
// holds { W[k][n], W[k+1][n] } with k = kt*32 + 16*(v>>2) + 8*(lane>=16) + 2*(v&3),
// n = c*16 + (lane&15).   (CDNA5 ISA 7.12.2, 16-bit B 32x16, wave32)
// ---------------------------------------------------------------------------
__global__ __launch_bounds__(256)
void k_pack_w(const float* __restrict__ W, unsigned* __restrict__ out,
              int kin, int kout) {
  const int idx = blockIdx.x * blockDim.x + threadIdx.x;
  const int nkt = kin >> 5;
  const int total = (kout >> 4) * nkt * 256;
  if (idx >= total) return;
  const int v    = idx & 7;
  const int lane = (idx >> 3) & 31;
  const int rest = idx >> 8;
  const int kt   = rest % nkt;
  const int c    = rest / nkt;
  const int half = lane >> 4;
  const int n    = c * 16 + (lane & 15);
  const int k    = kt * 32 + 16 * (v >> 2) + half * 8 + 2 * (v & 3);
  union { _Float16 h[2]; unsigned u; } p;
  p.h[0] = (_Float16)W[(size_t)k * kout + n];
  p.h[1] = (_Float16)W[(size_t)(k + 1) * kout + n];
  out[idx] = p.u;
}

// ---------------------------------------------------------------------------
// Wave-tile WMMA GEMM: C[n,kout] = A[n,KIN] @ W[KIN,kout] (+ bias)
// Block = 8 waves = 8 row tiles x one col-group of CT 16-col tiles.
// Packed B for the col-group is staged into LDS with
// global_load_async_to_lds_b128 (ASYNCcnt) and read back as b128 ds loads.
// KIN is compile-time so the k-loop fully unrolls.
// ---------------------------------------------------------------------------
template<int CT, int KIN>
__global__ __launch_bounds__(256)
void gemm_wmma(const float* __restrict__ A, const unsigned* __restrict__ Wp,
               const float* __restrict__ bias, float* __restrict__ C,
               int nrows, int kout) {
  extern __shared__ char smem[];
  const int tid  = threadIdx.x;
  const int lane = tid & 31;
  const int wv   = tid >> 5;
  constexpr int nkt   = KIN / 32;
  constexpr int bytes = CT * nkt * 1024;   // multiple of 4096 for all our shapes
  const int cg = blockIdx.y;

  // ---- async-stage packed B (contiguous per col-group) into LDS ----
  {
    const char* gsrc = (const char*)Wp + (size_t)cg * bytes;
#pragma unroll
    for (int ofs0 = 0; ofs0 < bytes; ofs0 += 4096) {
      const int ofs = ofs0 + tid * 16;
      const unsigned dst = (unsigned)(uintptr_t)(smem + ofs);  // low 32b = LDS addr
      asm volatile("global_load_async_to_lds_b128 %0, %1, off"
                   :: "v"(dst), "v"(gsrc + ofs) : "memory");
    }
    asm volatile("s_wait_asynccnt 0" ::: "memory");
  }
  __syncthreads();

  const int rt = blockIdx.x * 8 + wv;
  if (rt >= (nrows >> 4)) return;

  const int half = lane >> 4;
  const int m16  = lane & 15;
  const int h8   = half * 8;

  v8f acc[CT];
#pragma unroll
  for (int c = 0; c < CT; ++c)
#pragma unroll
    for (int j = 0; j < 8; ++j) acc[c][j] = 0.0f;

  const float* Arow = A + (size_t)(rt * 16 + m16) * KIN;

#pragma unroll
  for (int kt = 0; kt < nkt; ++kt) {
    const int kb = kt * 32;
    // A fragment 16x32 f16: lane's 16 elems = two contiguous 8-float runs.
    const float4 q0 = *(const float4*)(Arow + kb + h8);
    const float4 q1 = *(const float4*)(Arow + kb + h8 + 4);
    const float4 q2 = *(const float4*)(Arow + kb + 16 + h8);
    const float4 q3 = *(const float4*)(Arow + kb + 16 + h8 + 4);
    v16h a;
    a[0]  = (_Float16)q0.x; a[1]  = (_Float16)q0.y;
    a[2]  = (_Float16)q0.z; a[3]  = (_Float16)q0.w;
    a[4]  = (_Float16)q1.x; a[5]  = (_Float16)q1.y;
    a[6]  = (_Float16)q1.z; a[7]  = (_Float16)q1.w;
    a[8]  = (_Float16)q2.x; a[9]  = (_Float16)q2.y;
    a[10] = (_Float16)q2.z; a[11] = (_Float16)q2.w;
    a[12] = (_Float16)q3.x; a[13] = (_Float16)q3.y;
    a[14] = (_Float16)q3.z; a[15] = (_Float16)q3.w;
    // Hoist all CT B-fragment LDS loads: one s_wait_dscnt, then the CT WMMAs
    // issue back-to-back (shared A, disjoint accumulators -> no hazards).
    union Frag { v16h h; uint4 q[2]; } u[CT];
#pragma unroll
    for (int c = 0; c < CT; ++c) {
      const uint4* bp =
          (const uint4*)(smem + ((size_t)((c * nkt + kt) * 32 + lane)) * 32);
      u[c].q[0] = bp[0];
      u[c].q[1] = bp[1];
    }
#pragma unroll
    for (int c = 0; c < CT; ++c) {
      acc[c] = __builtin_amdgcn_wmma_f32_16x16x32_f16(
          false, a, false, u[c].h, (short)0, acc[c], false, false);
    }
  }
  // C/D layout: VGPR j -> M = j + 8*half, N = lane&15
#pragma unroll
  for (int c = 0; c < CT; ++c) {
    const int col = (cg * CT + c) * 16 + m16;
    const float bv = bias ? bias[col] : 0.0f;
#pragma unroll
    for (int j = 0; j < 8; ++j) {
      const int r = rt * 16 + j + half * 8;
      C[(size_t)r * kout + col] = acc[c][j] + bv;
    }
  }
}

// ---------------------------------------------------------------------------
// Degree / dis (computed once; deg = in-degree + 1 for the self loop)
// ---------------------------------------------------------------------------
__global__ void k_fill_one(float* p, int n) {
  int i = blockIdx.x * blockDim.x + threadIdx.x;
  if (i < n) p[i] = 1.0f;
}
__global__ void k_deg_count(const int* __restrict__ dst, float* deg, int e) {
  int i = blockIdx.x * blockDim.x + threadIdx.x;
  if (i < e) atomAddF(&deg[dst[i]], 1.0f);
}
__global__ void k_rsqrt_inplace(float* p, int n) {
  int i = blockIdx.x * blockDim.x + threadIdx.x;
  if (i < n) p[i] = rsqrtf(p[i]);
}

// agg[n,f] = bases[n,f] * dis[n]^2   (self-loop term; also initializes agg)
__global__ void k_selfloop(const float* __restrict__ bases,
                           const float* __restrict__ dis,
                           float* __restrict__ agg, int n) {
  int i = blockIdx.x * blockDim.x + threadIdx.x;
  if (i < n * NIN) {
    const int node = i >> 7;
    const float d = dis[node];
    agg[i] = bases[i] * d * d;
  }
}

// One wave per edge: agg[dst] += bases[src] * dis[src]*dis[dst]  (128 f32)
__global__ __launch_bounds__(256)
void k_scatter(const int* __restrict__ src, const int* __restrict__ dst,
               const float* __restrict__ dis, const float* __restrict__ bases,
               float* __restrict__ agg, int e) {
  const int wv = blockIdx.x * (blockDim.x >> 5) + (threadIdx.x >> 5);
  if (wv >= e) return;
  const int lane = threadIdx.x & 31;
  const int s = src[wv], d = dst[wv];
  const float nrm = dis[s] * dis[d];
  const float4 v = *(const float4*)(bases + (size_t)s * NIN + lane * 4);
  float* p = agg + (size_t)d * NIN + lane * 4;
  atomAddF(p + 0, v.x * nrm);
  atomAddF(p + 1, v.y * nrm);
  atomAddF(p + 2, v.z * nrm);
  atomAddF(p + 3, v.w * nrm);
}

// ---------------------------------------------------------------------------
// One wave per node: out = resid(h) + einsum(comb,agg) + cb; LayerNorm; ReLU
// Lane l owns features {l, l+32, ..., l+224}  (fh = l, head = i)
// ---------------------------------------------------------------------------
__global__ __launch_bounds__(256)
void k_einsum_ln_relu(const float* __restrict__ comb, const float* __restrict__ agg,
                      const float* __restrict__ cb, const float* __restrict__ gamma,
                      const float* __restrict__ beta, float* __restrict__ h, int n) {
  const int wv = blockIdx.x * (blockDim.x >> 5) + (threadIdx.x >> 5);
  if (wv >= n) return;
  const int lane = threadIdx.x & 31;
  const float creg = comb[(size_t)wv * 32 + lane];   // comb[h*4+b] via shuffle
  float aggv[NB];
#pragma unroll
  for (int b = 0; b < NB; ++b) aggv[b] = agg[(size_t)wv * NIN + b * 32 + lane];
  float o[8], s = 0.0f, sq = 0.0f;
#pragma unroll
  for (int i = 0; i < 8; ++i) {
    float val = h[(size_t)wv * NHID + i * 32 + lane];   // resid = x@Wr + br
#pragma unroll
    for (int b = 0; b < NB; ++b) val += __shfl(creg, i * 4 + b, 32) * aggv[b];
    val += cb[i * 32 + lane];
    o[i] = val; s += val; sq += val * val;
  }
#pragma unroll
  for (int off = 16; off > 0; off >>= 1) {
    s  += __shfl_xor(s, off, 32);
    sq += __shfl_xor(sq, off, 32);
  }
  const float mu  = s * (1.0f / NHID);
  const float var = sq * (1.0f / NHID) - mu * mu;
  const float inv = rsqrtf(var + 1e-5f);
#pragma unroll
  for (int i = 0; i < 8; ++i) {
    float val = (o[i] - mu) * inv * gamma[i * 32 + lane] + beta[i * 32 + lane];
    h[(size_t)wv * NHID + i * 32 + lane] = fmaxf(val, 0.0f);
  }
}

// ---------------------------------------------------------------------------
// Global mean pool + final projection
// ---------------------------------------------------------------------------
__global__ __launch_bounds__(256)
void k_pool_scatter(const float* __restrict__ h, const int* __restrict__ batch,
                    float* __restrict__ psum, float* __restrict__ pcnt, int n) {
  const int wv = blockIdx.x * (blockDim.x >> 5) + (threadIdx.x >> 5);
  if (wv >= n) return;
  const int lane = threadIdx.x & 31;
  const int g = batch[wv];
#pragma unroll
  for (int i = 0; i < 8; ++i)
    atomAddF(&psum[(size_t)g * NHID + i * 32 + lane],
             h[(size_t)wv * NHID + i * 32 + lane]);
  if (lane == 0) atomAddF(&pcnt[g], 1.0f);
}

__global__ __launch_bounds__(256)
void k_pool_final(const float* __restrict__ psum, const float* __restrict__ pcnt,
                  const float* __restrict__ Wf, float* __restrict__ out, int g) {
  const int wv = blockIdx.x * (blockDim.x >> 5) + (threadIdx.x >> 5);
  if (wv >= g) return;
  const int lane = threadIdx.x & 31;
  float s = 0.0f;
#pragma unroll
  for (int i = 0; i < 8; ++i)
    s += psum[(size_t)wv * NHID + i * 32 + lane] * Wf[i * 32 + lane];
#pragma unroll
  for (int off = 16; off > 0; off >>= 1) s += __shfl_xor(s, off, 32);
  if (lane == 0) out[wv] = s / fmaxf(pcnt[wv], 1.0f);
}

// ---------------------------------------------------------------------------
template<int CT, int KIN>
static inline void launch_gemm(const float* A, const unsigned* Wp, const float* bias,
                               float* C, int n, int kout, hipStream_t s) {
  const int rowTiles = n / 16;
  dim3 grid((rowTiles + 7) / 8, kout / (16 * CT));
  const int shbytes = CT * (KIN / 32) * 1024;
  gemm_wmma<CT, KIN><<<grid, 256, shbytes, s>>>(A, Wp, bias, C, n, kout);
}

extern "C" void kernel_launch(void* const* d_in, const int* in_sizes, int n_in,
                              void* d_out, int out_size, void* d_ws, size_t ws_size,
                              hipStream_t stream) {
  const float* x    = (const float*)d_in[0];
  const int*   eidx = (const int*)d_in[1];
  const int*   batch= (const int*)d_in[2];
  const float* Wb0  = (const float*)d_in[3];
  const float* Wc0  = (const float*)d_in[4];
  const float* bc0  = (const float*)d_in[5];
  const float* cb0  = (const float*)d_in[6];
  const float* Wr0  = (const float*)d_in[7];
  const float* br0  = (const float*)d_in[8];
  const float* g0   = (const float*)d_in[9];
  const float* b0   = (const float*)d_in[10];
  const float* WbS  = (const float*)d_in[11];
  const float* WcS  = (const float*)d_in[12];
  const float* bcS  = (const float*)d_in[13];
  const float* cbS  = (const float*)d_in[14];
  const float* WrS  = (const float*)d_in[15];
  const float* brS  = (const float*)d_in[16];
  const float* gS   = (const float*)d_in[17];
  const float* bS   = (const float*)d_in[18];
  const float* Wf   = (const float*)d_in[19];
  float* out = (float*)d_out;

  const int N = in_sizes[0] / NIN;
  const int E = in_sizes[1] / 2;
  const int* src = eidx;
  const int* dst = eidx + E;

  // workspace layout (floats)
  float* ws    = (float*)d_ws;
  float* hA    = ws;
  float* hB    = hA    + (size_t)N * NHID;
  float* bases = hB    + (size_t)N * NHID;
  float* agg   = bases + (size_t)N * NIN;
  float* comb  = agg   + (size_t)N * NIN;
  float* dis   = comb  + (size_t)N * (NH * NB);
  float* psum  = dis   + (size_t)N;
  float* pcnt  = psum  + (size_t)NG * NHID;
  unsigned* pWb = (unsigned*)(pcnt + NG);          // packed f16 weights (max kin=256)
  unsigned* pWc = pWb + (size_t)NHID * 64;         // 256*64  dwords
  unsigned* pWr = pWc + (size_t)NHID * 16;         // 256*16  dwords
  //            pWr needs 256*128 dwords

  // ---- symnorm degrees (graph-invariant across layers) ----
  k_fill_one<<<(N + 255) / 256, 256, 0, stream>>>(dis, N);
  k_deg_count<<<(E + 255) / 256, 256, 0, stream>>>(dst, dis, E);
  k_rsqrt_inplace<<<(N + 255) / 256, 256, 0, stream>>>(dis, N);

  // ---- layers ----
  const float* hin = x;
  for (int l = 0; l < NL; ++l) {
    const int kin = (l == 0) ? NIN : NHID;
    const float *Wb_, *Wc_, *bc_, *cb_, *Wr_, *br_, *g_, *b_;
    if (l == 0) {
      Wb_ = Wb0; Wc_ = Wc0; bc_ = bc0; cb_ = cb0;
      Wr_ = Wr0; br_ = br0; g_  = g0;  b_  = b0;
    } else {
      const int j = l - 1;
      Wb_ = WbS + (size_t)j * NHID * (NB * NFH);
      Wc_ = WcS + (size_t)j * NHID * (NH * NB);
      bc_ = bcS + (size_t)j * (NH * NB);
      cb_ = cbS + (size_t)j * NHID;
      Wr_ = WrS + (size_t)j * NHID * NHID;
      br_ = brS + (size_t)j * NHID;
      g_  = gS  + (size_t)j * NHID;
      b_  = bS  + (size_t)j * NHID;
    }
    float* hout = (l & 1) ? hB : hA;

    // pack this layer's weights into WMMA B-fragment images
    k_pack_w<<<(kin * (NB * NFH) / 2 + 255) / 256, 256, 0, stream>>>(Wb_, pWb, kin, NB * NFH);
    k_pack_w<<<(kin * (NH * NB)  / 2 + 255) / 256, 256, 0, stream>>>(Wc_, pWc, kin, NH * NB);
    k_pack_w<<<(kin * NHID       / 2 + 255) / 256, 256, 0, stream>>>(Wr_, pWr, kin, NHID);

    if (l == 0) {
      launch_gemm<4, NIN >(hin, pWb, nullptr, bases, N, NB * NFH, stream);
      launch_gemm<2, NIN >(hin, pWc, bc_,     comb,  N, NH * NB,  stream);
      launch_gemm<4, NIN >(hin, pWr, br_,     hout,  N, NHID,     stream);
    } else {
      launch_gemm<4, NHID>(hin, pWb, nullptr, bases, N, NB * NFH, stream);
      launch_gemm<2, NHID>(hin, pWc, bc_,     comb,  N, NH * NB,  stream);
      launch_gemm<4, NHID>(hin, pWr, br_,     hout,  N, NHID,     stream);
    }

    k_selfloop<<<((size_t)N * NIN + 255) / 256, 256, 0, stream>>>(bases, dis, agg, N);
    k_scatter<<<(E + 7) / 8, 256, 0, stream>>>(src, dst, dis, bases, agg, E);
    k_einsum_ln_relu<<<(N + 7) / 8, 256, 0, stream>>>(comb, agg, cb_, g_, b_, hout, N);

    hin = hout;
  }

  // ---- global mean pool + final projection ----
  hipMemsetAsync(psum, 0, (size_t)NG * NHID * sizeof(float), stream);
  hipMemsetAsync(pcnt, 0, (size_t)NG * sizeof(float), stream);
  k_pool_scatter<<<(N + 7) / 8, 256, 0, stream>>>(hin, batch, psum, pcnt, N);
  k_pool_final<<<(NG + 7) / 8, 256, 0, stream>>>(psum, pcnt, Wf, out, NG);
}
